// OptimizedCausalSelfAttention_52192442581222
// MI455X (gfx1250) — matrix operand order repaired and checked by reading, not compiled, not run on hardware
//
#include <hip/hip_runtime.h>
#include <hip/hip_bf16.h>

#define BB 4
#define TT 2048
#define DD 1024
#define HH 16
#define HDD 64
#define BT (BB*TT)

typedef __attribute__((ext_vector_type(16))) __bf16 v16bf;
typedef __attribute__((ext_vector_type(8)))  float  v8f;

union Frag16 { v16bf v; unsigned u[8]; unsigned short s[16]; };

__device__ __forceinline__ unsigned short f2bf(float f){
  union { float f; unsigned u; } x; x.f = f;
  unsigned u = x.u;
  u += 0x7FFFu + ((u >> 16) & 1u);   // round-to-nearest-even
  return (unsigned short)(u >> 16);
}

// A fragment: 16x32 (MxK) bf16, memory row-major with leading dim ldm.
// ISA layout: lane(0-15)=row M, vgpr i holds K pair {2i,2i+1}(+8 for i>=4), +8 for lanes 16-31.
__device__ __forceinline__ v16bf load_a16x32(const unsigned short* base, int ldm, int lane){
  const int half = lane >> 4, m = lane & 15;
  Frag16 f;
  const unsigned short* row = base + (size_t)m*ldm + half*8;
#pragma unroll
  for (int i=0;i<8;i++){
    const int kb = 2*i + ((i>=4)?8:0);
    f.u[i] = *(const unsigned*)(row + kb);
  }
  return f.v;
}

// B fragment: 32x16 (KxN) bf16 where memory holds element (k,n) at base[n*ldn + k]
// (K contiguous -> dword pairs). vgpr i holds K pair {2i,2i+1}(+8 for i>=4), +8 for lanes 16-31, col = lane&15.
__device__ __forceinline__ v16bf load_b32x16_kc(const unsigned short* base, size_t ldn, int lane){
  const int half = lane >> 4, n = lane & 15;
  Frag16 f;
  const unsigned short* col = base + (size_t)n*ldn + half*8;
#pragma unroll
  for (int i=0;i<8;i++) f.u[i] = *(const unsigned*)(col + 2*i + ((i>=4)?8:0));
  return f.v;
}

__global__ void cvt_f32_to_bf16(const float* __restrict__ in,
                                unsigned short* __restrict__ out, int n){
  int i  = blockIdx.x*blockDim.x + threadIdx.x;
  int st = gridDim.x*blockDim.x;
  for (; i < n; i += st) out[i] = f2bf(in[i]);
}

// Q/K/V projection: grid = (BT/128, H, 3). Wave computes 32 rows x 64 cols (one head strip).
// Q,K stored [B,H,T,HD]; V stored transposed [B,H,HD,T] so attention's P*V operand is dword-loadable.
__global__ __launch_bounds__(128) void qkv_wmma(
    const unsigned short* __restrict__ Xbf,
    const unsigned short* __restrict__ Wbf,
    const float* __restrict__ bq, const float* __restrict__ bk, const float* __restrict__ bv,
    unsigned short* __restrict__ QKV)
{
  const int lane = threadIdx.x & 31;
  const int wave = threadIdx.x >> 5;
  const int half = lane >> 4, nn = lane & 15;
  const int row0 = (blockIdx.x*4 + wave)*32;       // 32 token rows per wave
  const int h    = blockIdx.y;
  const int z    = blockIdx.z;                     // 0=Q 1=K 2=V

  const unsigned short* W = Wbf + (size_t)z*DD*DD;
  const float* bias = (z==0)? bq : (z==1)? bk : bv;
  unsigned short* out = QKV + (size_t)z*BT*DD;

  v8f acc[2][4] = {};
  for (int k = 0; k < DD; k += 32){
    v16bf aLo = load_a16x32(Xbf + (size_t)(row0     )*DD + k, DD, lane);
    v16bf aHi = load_a16x32(Xbf + (size_t)(row0 + 16)*DD + k, DD, lane);
#pragma unroll
    for (int j=0;j<4;j++){
      v16bf bf = load_b32x16_kc(W + (size_t)(h*HDD + j*16)*DD + k, DD, lane);
      acc[0][j] = __builtin_amdgcn_wmma_f32_16x16x32_bf16(false, aLo, false, bf,
                                                          (short)0, acc[0][j], false, false);
      acc[1][j] = __builtin_amdgcn_wmma_f32_16x16x32_bf16(false, aHi, false, bf,
                                                          (short)0, acc[1][j], false, false);
    }
  }
  const int b = row0 / TT, t0 = row0 % TT;
  const int bh = b*HH + h;
#pragma unroll
  for (int ti=0; ti<2; ti++){
#pragma unroll
    for (int j=0;j<4;j++){
      const float bia = bias[h*HDD + j*16 + nn];
#pragma unroll
      for (int r=0;r<8;r++){
        const int m = ti*16 + r + half*8;          // row within the 32-row strip
        const unsigned short val = f2bf(acc[ti][j][r] + bia);
        if (z == 2)                                 // V^T: [B,H,HD,T]
          out[((size_t)bh*HDD + (j*16 + nn))*TT + (t0 + m)] = val;
        else                                        // Q,K: [B,H,T,HD]
          out[((size_t)bh*TT + (t0 + m))*HDD + j*16 + nn] = val;
      }
    }
  }
}

// Flash attention: grid = (T/128, B*H), 4 waves/block, wave owns 32 query rows (2 A-tiles).
// Every K / V^T B-fragment is shared by both row tiles (2 WMMAs per fragment load).
__global__ __launch_bounds__(128) void attn_wmma(
    const unsigned short* __restrict__ Qb,
    const unsigned short* __restrict__ Kb,
    const unsigned short* __restrict__ Vtb,
    const float* __restrict__ mask,
    float* __restrict__ out)
{
  __shared__ unsigned short pbuf[4][2][16][32];    // per-wave, per-tile P staging (8 KB)
  const int lane = threadIdx.x & 31;
  const int wave = threadIdx.x >> 5;
  const int half = lane >> 4, nn = lane & 15;
  const int bh = blockIdx.y;
  const int b  = bh / HH, h = bh % HH;
  const int t0 = blockIdx.x*128 + wave*32;

  const unsigned short* q  = Qb  + (size_t)bh*TT*HDD;
  const unsigned short* k  = Kb  + (size_t)bh*TT*HDD;
  const unsigned short* vT = Vtb + (size_t)bh*HDD*TT;   // [HD][T]
  const float* mrow = mask + (size_t)b*TT;

  // Loop-invariant Q fragments: 2 row tiles x 2 K-chunks of HD
  v16bf aq[2][2];
#pragma unroll
  for (int ti=0; ti<2; ti++)
#pragma unroll
    for (int c=0; c<2; c++)
      aq[ti][c] = load_a16x32(q + (size_t)(t0 + ti*16)*HDD + c*32, HDD, lane);

  float mrun[2][8], lrun[2][8];
#pragma unroll
  for (int ti=0; ti<2; ti++)
#pragma unroll
    for (int r=0;r<8;r++){ mrun[ti][r] = -1e30f; lrun[ti][r] = 0.f; }
  v8f accY[2][4] = {};

  const float scale = 0.125f;                      // 1/sqrt(64)
  const int sEnd = t0 + 32;                        // causal bound for the 32-row strip
  for (int s0 = 0; s0 < sEnd; s0 += 32){
    if (s0 + 32 < sEnd)                            // prefetch next K chunk
      __builtin_prefetch(k + (size_t)(s0+32)*HDD, 0, 1);

    // K chunk fragments (shared by both row tiles)
    const v16bf b0 = load_b32x16_kc(k + (size_t)(s0   )*HDD +  0, HDD, lane);
    const v16bf b1 = load_b32x16_kc(k + (size_t)(s0   )*HDD + 32, HDD, lane);
    const v16bf b2 = load_b32x16_kc(k + (size_t)(s0+16)*HDD +  0, HDD, lane);
    const v16bf b3 = load_b32x16_kc(k + (size_t)(s0+16)*HDD + 32, HDD, lane);

#pragma unroll
    for (int ti=0; ti<2; ti++){
      v8f sc0 = {}, sc1 = {};
      sc0 = __builtin_amdgcn_wmma_f32_16x16x32_bf16(false, aq[ti][0], false, b0, (short)0, sc0, false, false);
      sc0 = __builtin_amdgcn_wmma_f32_16x16x32_bf16(false, aq[ti][1], false, b1, (short)0, sc0, false, false);
      sc1 = __builtin_amdgcn_wmma_f32_16x16x32_bf16(false, aq[ti][0], false, b2, (short)0, sc1, false, false);
      sc1 = __builtin_amdgcn_wmma_f32_16x16x32_bf16(false, aq[ti][1], false, b3, (short)0, sc1, false, false);

      // scale + additive mask + causal, in C layout (row = r+half*8, col = nn)
      const int tr0 = t0 + ti*16;
      float s0v[8], s1v[8];
#pragma unroll
      for (int r=0;r<8;r++){
        const int t = tr0 + r + half*8;
        const int c0 = s0 + nn, c1 = s0 + 16 + nn;
        float x0 = sc0[r]*scale + mrow[c0];
        float x1 = sc1[r]*scale + mrow[c1];
        s0v[r] = (c0 > t) ? -1e30f : x0;
        s1v[r] = (c1 > t) ? -1e30f : x1;
      }
      // online softmax; xor 1/2/4/8 reductions stay within each 16-lane half (= its 8 rows)
      float cmax[8];
#pragma unroll
      for (int r=0;r<8;r++) cmax[r] = fmaxf(s0v[r], s1v[r]);
#pragma unroll
      for (int off=1; off<16; off<<=1)
#pragma unroll
        for (int r=0;r<8;r++) cmax[r] = fmaxf(cmax[r], __shfl_xor(cmax[r], off, 32));
      float corr[8], p0[8], p1[8], rsum[8];
#pragma unroll
      for (int r=0;r<8;r++){
        const float mn = fmaxf(mrun[ti][r], cmax[r]);
        corr[r] = __expf(mrun[ti][r] - mn);
        mrun[ti][r] = mn;
        p0[r] = __expf(s0v[r] - mn);
        p1[r] = __expf(s1v[r] - mn);
        rsum[r] = p0[r] + p1[r];
      }
#pragma unroll
      for (int off=1; off<16; off<<=1)
#pragma unroll
        for (int r=0;r<8;r++) rsum[r] += __shfl_xor(rsum[r], off, 32);
#pragma unroll
      for (int r=0;r<8;r++) lrun[ti][r] = lrun[ti][r]*corr[r] + rsum[r];
#pragma unroll
      for (int j=0;j<4;j++)
#pragma unroll
        for (int r=0;r<8;r++) accY[ti][j][r] *= corr[r];

      // stage P through LDS (same-wave buffer, in-order DS)
#pragma unroll
      for (int r=0;r<8;r++){
        const int m = r + half*8;
        pbuf[wave][ti][m][nn]      = f2bf(p0[r]);
        pbuf[wave][ti][m][16 + nn] = f2bf(p1[r]);
      }
    }

    // P*V : V^T holds element (k=s, n=hd) at vT[hd*T + s] -> K-contiguous dword loads,
    // each V fragment feeds both row tiles.
    const v16bf pa0 = load_a16x32(&pbuf[wave][0][0][0], 32, lane);
    const v16bf pa1 = load_a16x32(&pbuf[wave][1][0][0], 32, lane);
#pragma unroll
    for (int j=0;j<4;j++){
      v16bf vb = load_b32x16_kc(vT + (size_t)(j*16)*TT + s0, TT, lane);
      accY[0][j] = __builtin_amdgcn_wmma_f32_16x16x32_bf16(false, pa0, false, vb,
                                                           (short)0, accY[0][j], false, false);
      accY[1][j] = __builtin_amdgcn_wmma_f32_16x16x32_bf16(false, pa1, false, vb,
                                                           (short)0, accY[1][j], false, false);
    }
  }
  // normalize and write f32 output [B,T,H*HD]
#pragma unroll
  for (int ti=0; ti<2; ti++){
#pragma unroll
    for (int r=0;r<8;r++){
      const int m = ti*16 + r + half*8;
      const float inv = 1.0f / lrun[ti][r];
      float* orow = out + ((size_t)b*TT + (t0+m))*DD + h*HDD;
#pragma unroll
      for (int j=0;j<4;j++) orow[j*16 + nn] = accY[ti][j][r] * inv;
    }
  }
}

extern "C" void kernel_launch(void* const* d_in, const int* in_sizes, int n_in,
                              void* d_out, int out_size, void* d_ws, size_t ws_size,
                              hipStream_t stream)
{
  const float* hs   = (const float*)d_in[0];
  const float* mask = (const float*)d_in[1];
  const float* Wq   = (const float*)d_in[2];
  const float* bq   = (const float*)d_in[3];
  const float* Wk   = (const float*)d_in[4];
  const float* bk   = (const float*)d_in[5];
  const float* Wv   = (const float*)d_in[6];
  const float* bv   = (const float*)d_in[7];
  float* out = (float*)d_out;

  // workspace: Xbf [BT*D] | Wbf [3*D*D] | QKV bf16 [3 * BT*D]  (V region holds V^T)
  unsigned short* Xbf = (unsigned short*)d_ws;
  unsigned short* Wbf = Xbf + (size_t)BT*DD;
  unsigned short* QKV = Wbf + (size_t)3*DD*DD;

  cvt_f32_to_bf16<<<1024, 256, 0, stream>>>(hs, Xbf, BT*DD);
  cvt_f32_to_bf16<<<512, 256, 0, stream>>>(Wq, Wbf + 0*(size_t)DD*DD, DD*DD);
  cvt_f32_to_bf16<<<512, 256, 0, stream>>>(Wk, Wbf + 1*(size_t)DD*DD, DD*DD);
  cvt_f32_to_bf16<<<512, 256, 0, stream>>>(Wv, Wbf + 2*(size_t)DD*DD, DD*DD);

  dim3 gq(BT/128, HH, 3);                // 64 x 16 x 3 blocks, 4 waves each
  qkv_wmma<<<gq, 128, 0, stream>>>(Xbf, Wbf, bq, bk, bv, QKV);

  dim3 ga(TT/128, BB*HH);                // 16 x 64 blocks, 4 waves each
  attn_wmma<<<ga, 128, 0, stream>>>(QKV,
                                    QKV + (size_t)BT*DD,
                                    QKV + 2*(size_t)BT*DD,
                                    mask, out);
}
